// SelectiveSSM_6725918785939
// MI455X (gfx1250) — matrix operand order repaired and checked by reading, not compiled, not run on hardware
//
#include <hip/hip_runtime.h>
#include <hip/hip_bf16.h>
#include <stdint.h>

// ---------------- problem constants ----------------
#define D_MODEL 1024
#define D_STATE 16
#define D_CONV  4
#define D_INNER 2048            // 2*D_MODEL
#define BATCH   2
#define SEQ     2048
#define NTOK    (BATCH*SEQ)     // 4096 tokens
#define NX      (2*D_INNER)     // 4096 = W_in output cols
#define NP      (2*D_STATE + D_INNER) // 2080 = W_x output cols (logical)
#define NPP     2176            // NP padded up to a multiple of 128 for the GEMM

typedef __attribute__((ext_vector_type(16))) __bf16 v16bf;
typedef __attribute__((ext_vector_type(8)))  float  v8f;
typedef __attribute__((ext_vector_type(4)))  unsigned int u32x4;
typedef __attribute__((ext_vector_type(8)))  int i32x8;
typedef __attribute__((ext_vector_type(4)))  int i32x4;

#if defined(__has_builtin)
#if __has_builtin(__builtin_amdgcn_tensor_load_to_lds) && __has_builtin(__builtin_amdgcn_s_wait_tensorcnt)
#define HAVE_TDM 1
#endif
#endif
#ifndef HAVE_TDM
#define HAVE_TDM 0
#endif

// ---------------- elementwise converts ----------------
__global__ void __launch_bounds__(256) k_f32_to_bf16(const float* __restrict__ in,
                                                     __bf16* __restrict__ out, int n) {
  int i = blockIdx.x * 256 + threadIdx.x;
  if (i < n) out[i] = (__bf16)in[i];
}

// in: rows x cols row-major (fp32)  ->  out: cols x rows row-major (bf16)
__global__ void __launch_bounds__(256) k_transpose_bf16(const float* __restrict__ in,
                                                        __bf16* __restrict__ out,
                                                        int rows, int cols) {
  int i = blockIdx.x * 256 + threadIdx.x;
  if (i >= rows * cols) return;
  int n = i / rows;   // output row  == original col
  int k = i % rows;   // output col  == original row
  out[(size_t)n * rows + k] = (__bf16)in[(size_t)k * cols + n];
}

// ---------------- WMMA GEMM with TDM (Tensor Data Mover) staging ----------------
// C_f32[M,N] = A_bf16[M,K] * B, with B supplied transposed as Bt[N,K] (row-major).
// Block tile 128x128, K-step 32. 256 threads = 8 waves in a 4(M) x 2(N) grid;
// each wave owns a 32x64 register tile = 2x4 WMMA accumulators (8 WMMA / K-step).
// A/B tiles staged into double-buffered LDS by the TDM: one 2-D descriptor per
// tile (tile 32 x 128 elems, data_size=2B, row stride K), with TDM's hardware
// LDS padding (4 DWORDs every 16 DWORDs) producing an 80B row pitch: bank
// stride 20 (mod 64) -> conflict-free row-strided ds_load_b128 fragment reads.
#define BM 128
#define BN 128
#define BK 32
#define LDSW 40   // bf16 elements per LDS row (32 data + 8 pad) = 80B pitch

#if HAVE_TDM
// Build D# groups per CDNA5 ISA ch.8 and issue TENSOR_LOAD_TO_LDS.
// gaddr = global address of tile start (row-major, row pitch = K bf16 elems);
// lds_off = LDS byte offset of the destination tile.
__device__ __forceinline__ void tdm_load_tile(const __bf16* gaddr, unsigned int lds_off, int K) {
  unsigned long long ga = (unsigned long long)(uintptr_t)gaddr;
  u32x4 g0;
  g0[0] = 1u;                                   // count=1, user-mode, no gather
  g0[1] = lds_off;                              // lds_addr [63:32]
  g0[2] = (unsigned int)ga;                     // global_addr [95:64]
  g0[3] = (unsigned int)((ga >> 32) & 0x1FFFFFFu) | (2u << 30); // addr[56:32], type=2

  const unsigned int tensor_d0 = (unsigned int)K;  // dim0 length (elems)
  const unsigned int tensor_d1 = 1u << 20;         // dim1 length (rows), generous
  const unsigned int tile_d0 = BK;                 // 32 elems per row
  const unsigned int tile_d1 = BM;                 // 128 rows
  const unsigned long long stride0 = (unsigned long long)(unsigned int)K; // elems

  i32x8 g1;
  g1[0] = (int)((1u << 16)        // data_size = 1 -> 2-byte elements
              | (1u << 20)        // pad_enable
              | (3u << 22)        // pad_interval: 16 DWORDs (64B) between pads
              | (3u << 25));      // pad_amount:   4 DWORDs (16B) -> 80B pitch
  g1[1] = (int)((tensor_d0 & 0xFFFFu) << 16);                        // [63:48] = dim0 lo
  g1[2] = (int)((tensor_d0 >> 16) | ((tensor_d1 & 0xFFFFu) << 16));  // dim0 hi | dim1 lo
  g1[3] = (int)((tensor_d1 >> 16) | (tile_d0 << 16));                // dim1 hi | tile_dim0
  g1[4] = (int)(tile_d1);                                            // tile_dim1 | tile_dim2=0
  g1[5] = (int)(unsigned int)(stride0 & 0xFFFFFFFFu);                // dim0_stride lo
  g1[6] = (int)(unsigned int)((stride0 >> 32) & 0xFFFFu);            // dim0_stride hi
  g1[7] = 0;                                                         // dim1_stride (unused, 2-D)

  i32x4 z4 = {0, 0, 0, 0};
#if __clang_major__ >= 23
  i32x8 z8 = {0, 0, 0, 0, 0, 0, 0, 0};
  __builtin_amdgcn_tensor_load_to_lds(g0, g1, z4, z4, z8, 0);
#else
  __builtin_amdgcn_tensor_load_to_lds(g0, g1, z4, z4, 0);
#endif
}
#endif

__global__ void __launch_bounds__(256) k_gemm_bf16_wmma(const __bf16* __restrict__ A,
                                                        const __bf16* __restrict__ Bt,
                                                        float* __restrict__ C,
                                                        int ldc, int K) {
  const int tid  = threadIdx.x;
  const int wave = tid >> 5;
  const int lane = tid & 31;
  const int half = lane >> 4;      // lane half selects K sub-ranges per ISA layout
  const int r    = lane & 15;
  const int wm   = wave >> 1;      // 0..3 : which 32-row slice of the block tile
  const int wn   = wave & 1;       // 0..1 : which 64-col slice of the block tile

  const size_t blockM = (size_t)blockIdx.y * BM;
  const size_t blockN = (size_t)blockIdx.x * BN;

  v8f acc[2][4];
#pragma unroll
  for (int i = 0; i < 2; ++i)
#pragma unroll
    for (int j = 0; j < 4; ++j) acc[i][j] = (v8f){};

#if HAVE_TDM
  __shared__ __bf16 lA[2][BM * LDSW];
  __shared__ __bf16 lB[2][BN * LDSW];

  const __bf16* Abase = A  + blockM * K;
  const __bf16* Bbase = Bt + blockN * K;
  const int nk = K / BK;

  if (wave == 0) {   // one DMA pair per buffer; TENSORcnt tracks completion
    tdm_load_tile(Abase, (unsigned int)(uintptr_t)&lA[0][0], K);
    tdm_load_tile(Bbase, (unsigned int)(uintptr_t)&lB[0][0], K);
  }

  for (int it = 0; it < nk; ++it) {
    const int cur = it & 1;
    if (wave == 0) {
      if (it + 1 < nk) {   // prefetch next K-slab into the other buffer
        tdm_load_tile(Abase + (size_t)(it + 1) * BK, (unsigned int)(uintptr_t)&lA[cur ^ 1][0], K);
        tdm_load_tile(Bbase + (size_t)(it + 1) * BK, (unsigned int)(uintptr_t)&lB[cur ^ 1][0], K);
        __builtin_amdgcn_s_wait_tensorcnt(2);  // current pair done; next pair in flight
      } else {
        __builtin_amdgcn_s_wait_tensorcnt(0);
      }
    }
    __syncthreads();   // publish TDM-written LDS to all waves

    union Frag { uint4 q[2]; v16bf h; } fa[2], fb[4];
#pragma unroll
    for (int i = 0; i < 2; ++i) {
      const __bf16* ar = &lA[cur][(wm * 32 + i * 16 + r) * LDSW];
      fa[i].q[0] = *(const uint4*)(ar + half * 8);        // K 0..7   / 8..15
      fa[i].q[1] = *(const uint4*)(ar + 16 + half * 8);   // K 16..23 / 24..31
    }
#pragma unroll
    for (int j = 0; j < 4; ++j) {
      const __bf16* br = &lB[cur][(wn * 64 + j * 16 + r) * LDSW];
      fb[j].q[0] = *(const uint4*)(br + half * 8);        // K 0..7   / 8..15
      fb[j].q[1] = *(const uint4*)(br + 16 + half * 8);   // K 16..23 / 24..31
    }

#pragma unroll
    for (int i = 0; i < 2; ++i)
#pragma unroll
      for (int j = 0; j < 4; ++j)
        acc[i][j] = __builtin_amdgcn_wmma_f32_16x16x32_bf16(
            false, fa[i].h, false, fb[j].h, (short)0, acc[i][j], false, false);

    __syncthreads();   // all reads of buf[cur] done before wave 0 reloads it
  }
#else
  // -------- fallback: manual cooperative staging --------
  __shared__ __bf16 lA1[BM * LDSW];
  __shared__ __bf16 lB1[BN * LDSW];
  const int c0row = (tid * 2) >> 2, c0ch = (tid * 2) & 3;
  const int c1row = (tid * 2 + 1) >> 2, c1ch = (tid * 2 + 1) & 3;

  for (int kb = 0; kb < K; kb += BK) {
    __syncthreads();
    {
      uint4 va0 = *(const uint4*)(A  + (blockM + c0row) * K + kb + c0ch * 8);
      uint4 va1 = *(const uint4*)(A  + (blockM + c1row) * K + kb + c1ch * 8);
      uint4 vb0 = *(const uint4*)(Bt + (blockN + c0row) * K + kb + c0ch * 8);
      uint4 vb1 = *(const uint4*)(Bt + (blockN + c1row) * K + kb + c1ch * 8);
      *(uint4*)(lA1 + c0row * LDSW + c0ch * 8) = va0;
      *(uint4*)(lA1 + c1row * LDSW + c1ch * 8) = va1;
      *(uint4*)(lB1 + c0row * LDSW + c0ch * 8) = vb0;
      *(uint4*)(lB1 + c1row * LDSW + c1ch * 8) = vb1;
    }
    __syncthreads();

    union Frag { uint4 q[2]; v16bf h; } fa[2], fb[4];
#pragma unroll
    for (int i = 0; i < 2; ++i) {
      const __bf16* ar = lA1 + (wm * 32 + i * 16 + r) * LDSW;
      fa[i].q[0] = *(const uint4*)(ar + half * 8);
      fa[i].q[1] = *(const uint4*)(ar + 16 + half * 8);
    }
#pragma unroll
    for (int j = 0; j < 4; ++j) {
      const __bf16* br = lB1 + (wn * 64 + j * 16 + r) * LDSW;
      fb[j].q[0] = *(const uint4*)(br + half * 8);
      fb[j].q[1] = *(const uint4*)(br + 16 + half * 8);
    }
#pragma unroll
    for (int i = 0; i < 2; ++i)
#pragma unroll
      for (int j = 0; j < 4; ++j)
        acc[i][j] = __builtin_amdgcn_wmma_f32_16x16x32_bf16(
            false, fa[i].h, false, fb[j].h, (short)0, acc[i][j], false, false);
  }
#endif

  // ---- epilogue: C/D layout -> row M = g + 8*half, col N = r ----
#pragma unroll
  for (int i = 0; i < 2; ++i)
#pragma unroll
    for (int j = 0; j < 4; ++j) {
      float* Cp = C + (blockM + wm * 32 + i * 16) * ldc + blockN + wn * 64 + j * 16 + r;
#pragma unroll
      for (int g = 0; g < 8; ++g)
        Cp[(size_t)(g + half * 8) * ldc] = acc[i][j][g];
    }
}

// ---------------- depthwise causal conv (width 4) + SiLU ----------------
__global__ void __launch_bounds__(256) k_conv_silu(const float* __restrict__ xr,
                                                   const float* __restrict__ cw,
                                                   const float* __restrict__ cb,
                                                   float* __restrict__ xc32,
                                                   __bf16* __restrict__ xcbf) {
  int i = blockIdx.x * 256 + threadIdx.x;
  if (i >= NTOK * D_INNER) return;
  int c = i % D_INNER;
  int g = i / D_INNER;
  int t = g % SEQ;
  float acc = cb[c];
#pragma unroll
  for (int j = 0; j < D_CONV; ++j) {
    int tt = t - (D_CONV - 1) + j;
    if (tt >= 0)
      acc += xr[(size_t)(g - (D_CONV - 1) + j) * NX + c] * cw[c * D_CONV + j];
  }
  float s = acc / (1.0f + __expf(-acc));   // SiLU
  xc32[i] = s;
  xcbf[i] = (__bf16)s;
}

// ---------------- silu(res), res = xr[:, D_INNER:] ----------------
__global__ void __launch_bounds__(256) k_silu_res(const float* __restrict__ xr,
                                                  float* __restrict__ sres) {
  int i = blockIdx.x * 256 + threadIdx.x;
  if (i >= NTOK * D_INNER) return;
  int c = i % D_INNER;
  int g = i / D_INNER;
  float v = xr[(size_t)g * NX + D_INNER + c];
  sres[i] = v / (1.0f + __expf(-v));
}

// ---------------- delta = softplus(p[:, :16] @ W_dt + b_dt)  (K=16, VALU) ----------------
__global__ void __launch_bounds__(256) k_delta(const float* __restrict__ p,
                                               const float* __restrict__ Wdt,
                                               const float* __restrict__ bdt,
                                               float* __restrict__ delta) {
  int i = blockIdx.x * 256 + threadIdx.x;
  if (i >= NTOK * D_INNER) return;
  int c = i % D_INNER;
  int g = i / D_INNER;
  const float* pr = p + (size_t)g * NPP;
  float acc = bdt[c];
#pragma unroll
  for (int j = 0; j < D_STATE; ++j)
    acc += pr[j] * Wdt[j * D_INNER + c];
  delta[i] = (acc > 20.0f) ? acc : log1pf(__expf(acc));
}

// ---------------- sequential selective scan: one thread per (batch, channel) ----------------
__global__ void __launch_bounds__(256) k_scan(const float* __restrict__ xc32,
                                              const float* __restrict__ delta,
                                              const float* __restrict__ p,
                                              const float* __restrict__ sres,
                                              __bf16* __restrict__ ybf) {
  int c = blockIdx.x * 256 + threadIdx.x;   // channel
  int b = blockIdx.y;                       // batch
  float h[D_STATE];
#pragma unroll
  for (int n = 0; n < D_STATE; ++n) h[n] = 0.0f;

  for (int t = 0; t < SEQ; ++t) {
    size_t g = (size_t)b * SEQ + t;
    float x  = xc32[g * D_INNER + c];
    float d  = delta[g * D_INNER + c];
    float Cc = p[g * NPP + 2 * D_STATE + c];
    const float* Bp = p + g * NPP + D_STATE;   // uniform across lanes -> scalar loads
    float sum = 0.0f;
#pragma unroll
    for (int n = 0; n < D_STATE; ++n) {
      h[n] = h[n] * d + Bp[n] * x;
      sum += h[n];
    }
    ybf[g * D_INNER + c] = (__bf16)(sum * Cc * sres[g * D_INNER + c]);
  }
}

// ---------------- host-side launch ----------------
static inline size_t align256(size_t x) { return (x + 255) & ~(size_t)255; }

extern "C" void kernel_launch(void* const* d_in, const int* in_sizes, int n_in,
                              void* d_out, int out_size, void* d_ws, size_t ws_size,
                              hipStream_t stream) {
  const float* x      = (const float*)d_in[0];  // (2,2048,1024)
  const float* W_in   = (const float*)d_in[1];  // (1024,4096)
  const float* conv_w = (const float*)d_in[2];  // (2048,1,4)
  const float* conv_b = (const float*)d_in[3];  // (2048,)
  const float* W_x    = (const float*)d_in[4];  // (2048,2080)
  const float* W_dt   = (const float*)d_in[5];  // (16,2048)
  const float* b_dt   = (const float*)d_in[6];  // (2048,)
  const float* W_out  = (const float*)d_in[7];  // (2048,1024)
  float* out = (float*)d_out;                   // (2,2048,1024) fp32

  char* ws = (char*)d_ws;
  size_t off = 0;
  auto carve = [&](size_t bytes) { void* p = ws + off; off += align256(bytes); return p; };

  __bf16* xbf   = (__bf16*)carve((size_t)NTOK * D_MODEL * 2);   // 4096x1024 bf16
  __bf16* WinT  = (__bf16*)carve((size_t)NX * D_MODEL * 2);     // 4096x1024 bf16 (W_in^T)
  __bf16* WxT   = (__bf16*)carve((size_t)NPP * D_INNER * 2);    // 2176x2048 bf16 (W_x^T, padded)
  __bf16* WoutT = (__bf16*)carve((size_t)D_MODEL * D_INNER * 2);// 1024x2048 bf16 (W_out^T)
  float*  xr    = (float*)carve((size_t)NTOK * NX * 4);         // 4096x4096 f32
  float*  xc32  = (float*)carve((size_t)NTOK * D_INNER * 4);    // 4096x2048 f32
  __bf16* xcbf  = (__bf16*)carve((size_t)NTOK * D_INNER * 2);   // 4096x2048 bf16
  float*  p     = (float*)carve((size_t)NTOK * NPP * 4);        // 4096x2176 f32 (padded)
  float*  dl    = (float*)carve((size_t)NTOK * D_INNER * 4);    // 4096x2048 f32
  float*  sres  = (float*)carve((size_t)NTOK * D_INNER * 4);    // 4096x2048 f32
  __bf16* ybf   = (__bf16*)carve((size_t)NTOK * D_INNER * 2);   // 4096x2048 bf16
  (void)ws_size; (void)in_sizes; (void)n_in; (void)out_size;

  // 1) converts / weight transposes to bf16
  k_f32_to_bf16<<<(NTOK * D_MODEL) / 256, 256, 0, stream>>>(x, xbf, NTOK * D_MODEL);
  k_transpose_bf16<<<(D_MODEL * NX) / 256, 256, 0, stream>>>(W_in, WinT, D_MODEL, NX);
  k_transpose_bf16<<<(D_INNER * NP + 255) / 256, 256, 0, stream>>>(W_x, WxT, D_INNER, NP);
  k_transpose_bf16<<<(D_INNER * D_MODEL) / 256, 256, 0, stream>>>(W_out, WoutT, D_INNER, D_MODEL);

  // 2) xr = x @ W_in           (M=4096, N=4096, K=1024)
  k_gemm_bf16_wmma<<<dim3(NX / BN, NTOK / BM), 256, 0, stream>>>(xbf, WinT, xr, NX, D_MODEL);

  // 3) conv + SiLU ; SiLU(res)
  k_conv_silu<<<(NTOK * D_INNER) / 256, 256, 0, stream>>>(xr, conv_w, conv_b, xc32, xcbf);
  k_silu_res<<<(NTOK * D_INNER) / 256, 256, 0, stream>>>(xr, sres);

  // 4) p = xc @ W_x            (M=4096, N=2176 padded, K=2048); cols >= 2080 are scratch
  k_gemm_bf16_wmma<<<dim3(NPP / BN, NTOK / BM), 256, 0, stream>>>(xcbf, WxT, p, NPP, D_INNER);

  // 5) delta = softplus(p[:, :16] @ W_dt + b_dt)
  k_delta<<<(NTOK * D_INNER) / 256, 256, 0, stream>>>(p, W_dt, b_dt, dl);

  // 6) sequential scan -> y (bf16), fused with * silu(res)
  dim3 sg(D_INNER / 256, BATCH);
  k_scan<<<sg, 256, 0, stream>>>(xc32, dl, p, sres, ybf);

  // 7) out = y @ W_out         (M=4096, N=1024, K=2048) -> fp32 d_out
  k_gemm_bf16_wmma<<<dim3(D_MODEL / BN, NTOK / BM), 256, 0, stream>>>(ybf, WoutT, out, D_MODEL, D_INNER);
}
